// RecurrentGCN_76295799046308
// MI455X (gfx1250) — compile-verified, matrix operand order checked by reading, not run on hardware
//
#include <hip/hip_runtime.h>

// ---------------------------------------------------------------------------
// MI455X (gfx1250) DCRNN forward.
//  - Dense [N x cin] x [cin x 64] GEMMs via v_wmma_f32_16x16x32_bf16
//    (bf16 inputs, fp32 accumulate; wave32, 16x16 tiles per ISA layout).
//  - Node-feature buffers padded to cinp = ceil(cin/32)*32 (66->96, 128->128):
//    zero extra WMMA k-steps, fully unconditional vectorized staging.
//  - Diffusion SpMM via edge-parallel fp32 global atomics (operands L2-resident:
//    N*cinp*4B <= 51MB << 192MB L2).
//  - Chebyshev terms shared between Z and R gates (halves SpMM traffic).
// ---------------------------------------------------------------------------

typedef __bf16 bf16_t;
typedef __attribute__((ext_vector_type(16))) __bf16 v16bf;
typedef __attribute__((ext_vector_type(8)))  __bf16 v8bf;
typedef __attribute__((ext_vector_type(8)))  float  v8f;

#define NN 100000   // nodes
#define HIDDEN 64
#define GEMM_ROWS 128
#define LDA 40      // bf16 units per LDS row (32 data + 8 pad, 16B aligned)
#define LDW 40

__device__ __forceinline__ unsigned short bfbits(float f) {
  union { float f; unsigned u; } in; in.f = f;
  unsigned u = in.u;
  unsigned r = u + 0x7FFFu + ((u >> 16) & 1u);   // round-to-nearest-even
  return (unsigned short)(r >> 16);
}
__device__ __forceinline__ bf16_t f2bf(float f) {
  union { unsigned short s; bf16_t b; } out;
  out.s = bfbits(f);
  return out.b;
}

// ------------------------------ utility kernels ----------------------------

__global__ void zero_kernel(float* p, long long n) {
  long long i = (long long)blockIdx.x * blockDim.x + threadIdx.x;
  if (i < n) p[i] = 0.f;
}

__global__ void degree_kernel(const float* __restrict__ w,
                              const int* __restrict__ src,
                              const int* __restrict__ dst,
                              float* dout, float* din, int E) {
  int e = blockIdx.x * blockDim.x + threadIdx.x;
  if (e >= E) return;
  float we = w[e];
  atomicAdd(&dout[src[e]], we);
  atomicAdd(&din[dst[e]], we);
}

__global__ void invert_kernel(float* p, int n) {
  int i = blockIdx.x * blockDim.x + threadIdx.x;
  if (i >= n) return;
  float v = p[i];
  p[i] = (v > 0.f) ? (1.f / v) : 0.f;
}

__global__ void extract_xs_kernel(const float* __restrict__ x, float* __restrict__ xs,
                                  long long n2) {
  long long i = (long long)blockIdx.x * blockDim.x + threadIdx.x;
  if (i >= n2) return;
  xs[i] = x[(i >> 1) * 24 + (i & 1)];   // x[:, :, 0, :] of [4,25000,12,2]
}

// pack W [3][2][3][cin][64] fp32 -> [3 gates][5 terms][64 cols][cinp k] bf16,
// term0 = W[0,0]+W[1,0]; k >= cin zero-filled.
__global__ void pack_weights_kernel(const float* __restrict__ W, bf16_t* __restrict__ out,
                                    int cin, int cinp, long long total) {
  long long idx = (long long)blockIdx.x * blockDim.x + threadIdx.x;
  if (idx >= total) return;
  long long per_gate = 5LL * 64 * cinp;
  int g = (int)(idx / per_gate);
  long long rem = idx % per_gate;
  int t  = (int)(rem / (64LL * cinp));
  long long rem2 = rem % (64LL * cinp);
  int o  = (int)(rem2 / cinp);   // output column
  int cc = (int)(rem2 % cinp);   // k index
  float v = 0.f;
  if (cc < cin) {
    const float* Wg = W + (long long)g * 2 * 3 * cin * 64;
    long long co = (long long)cc * 64 + o;
    long long dk00 = 0, dk10 = 3LL * cin * 64;
    switch (t) {
      case 0: v = Wg[dk00 + co] + Wg[dk10 + co]; break;                 // W[0,0]+W[1,0]
      case 1: v = Wg[dk00 + 1LL * cin * 64 + co]; break;                // W[0,1]
      case 2: v = Wg[dk10 + 1LL * cin * 64 + co]; break;                // W[1,1]
      case 3: v = Wg[dk00 + 2LL * cin * 64 + co]; break;                // W[0,2]
      default: v = Wg[dk10 + 2LL * cin * 64 + co]; break;               // W[1,2]
    }
  }
  out[idx] = f2bf(v);
}

// v = [x | h | 0pad]   or   v = [x | sigmoid(gateH) * h | 0pad]
__global__ void concat_kernel(float* __restrict__ v, const float* __restrict__ x, int F,
                              const float* __restrict__ h, const float* __restrict__ gateH,
                              long long total, int cinp) {
  long long idx = (long long)blockIdx.x * blockDim.x + threadIdx.x;
  if (idx >= total) return;
  long long i = idx / cinp;
  int j = (int)(idx % cinp);
  float val = 0.f;
  if (j < F) {
    val = x[i * F + j];
  } else if (j < F + HIDDEN) {
    float hv = h ? h[i * HIDDEN + (j - F)] : 0.f;
    if (gateH) {
      float g = 1.f / (1.f + __expf(-gateH[i * HIDDEN + (j - F)]));
      hv *= g;
    }
    val = hv;
  }
  v[idx] = val;
}

__global__ void scale_kernel(float* __restrict__ s, const float* __restrict__ v,
                             const float* __restrict__ dinv, long long total, int cinp) {
  long long idx = (long long)blockIdx.x * blockDim.x + threadIdx.x;
  if (idx >= total) return;
  s[idx] = dinv[idx / cinp] * v[idx];
}

// out[to] += s[from]  (node-normalized gather/scatter), float2 over real cin,
// rows strided by cinp (pad columns untouched, stay zero).
__global__ void spmm_kernel(const float* __restrict__ s, const int* __restrict__ from,
                            const int* __restrict__ to, float* __restrict__ out,
                            int cin, int cinp, long long total) {
  long long idx = (long long)blockIdx.x * blockDim.x + threadIdx.x;
  if (idx >= total) return;
  int cin2 = cin >> 1;
  long long e = idx / cin2;
  int q = (int)(idx % cin2);
  int f = from[e], tn = to[e];
  const float2 val = *(const float2*)(s + (size_t)f * cinp + 2 * q);
  float* o = out + (size_t)tn * cinp + 2 * q;
  atomicAdd(o, val.x);
  atomicAdd(o + 1, val.y);
}

__global__ void axpby_kernel(float* __restrict__ t, const float* __restrict__ v,
                             long long total) {   // t = 2*t - v  (Chebyshev)
  long long idx = (long long)blockIdx.x * blockDim.x + threadIdx.x;
  if (idx >= total) return;
  t[idx] = 2.f * t[idx] - v[idx];
}

__global__ void sigmoid_kernel(float* __restrict__ p, long long n) {
  long long i = (long long)blockIdx.x * blockDim.x + threadIdx.x;
  if (i >= n) return;
  p[i] = 1.f / (1.f + __expf(-p[i]));
}

__global__ void relu_kernel(const float* __restrict__ in, float* __restrict__ out, long long n) {
  long long i = (long long)blockIdx.x * blockDim.x + threadIdx.x;
  if (i >= n) return;
  float v = in[i];
  out[i] = v > 0.f ? v : 0.f;
}

__global__ void gru_kernel(float* __restrict__ hout, const float* __restrict__ Z,
                           const float* __restrict__ hprev, const float* __restrict__ Hc,
                           long long n) {
  long long i = (long long)blockIdx.x * blockDim.x + threadIdx.x;
  if (i >= n) return;
  float z = Z[i];
  float hp = hprev ? hprev[i] : 0.f;
  hout[i] = z * hp + (1.f - z) * tanhf(Hc[i]);
}

__global__ void linear_kernel(const float* __restrict__ h, const float* __restrict__ W,
                              const float* __restrict__ b, float* __restrict__ out, int N) {
  int i = blockIdx.x * blockDim.x + threadIdx.x;
  if (i >= N) return;
  float a0 = b[0], a1 = b[1];
  const float* hr = h + (size_t)i * HIDDEN;
  #pragma unroll 8
  for (int j = 0; j < HIDDEN; ++j) {
    float hv = hr[j];
    a0 += hv * W[j * 2 + 0];
    a1 += hv * W[j * 2 + 1];
  }
  out[i * 2 + 0] = a0;
  out[i * 2 + 1] = a1;
}

// ------------------------------ WMMA GEMM ----------------------------------
// H[N x 64] (accum or =) A[N x cinp(padded)] * W[64 x cinp, col-major terms] (+ bias).
// 256 thr = 8 wave32; block tile 128x64; wave w owns rows [16w,16w+16).
// A/B fragment layout per CDNA5 ISA 16-bit 16x32 A-matrix table:
//   lane l: m = l&15, base = (l<16 ? 0 : 8); elem i<8 -> K=base+i, i>=8 -> K=16+base+(i-8)
__global__ __launch_bounds__(256)
void gemm_wmma(const float* __restrict__ A, const bf16_t* __restrict__ W,
               float* __restrict__ H, const float* __restrict__ bias,
               int N, int cinp, int accumulate) {
  __shared__ bf16_t sA[GEMM_ROWS * LDA];
  __shared__ bf16_t sW[64 * LDW];          // transposed: [col][k]
  const int tid  = threadIdx.x;
  const int lane = tid & 31;
  const int wave = tid >> 5;
  const int row0 = blockIdx.x * GEMM_ROWS;

  v8f acc[4];
  #pragma unroll
  for (int t = 0; t < 4; ++t)
    #pragma unroll
    for (int r = 0; r < 8; ++r) acc[t][r] = 0.f;

  const int m    = (wave << 4) + (lane & 15);
  const int base = (lane < 16) ? 0 : 8;
  // W staging assignment: 64 cols x 4 chunks of 8 bf16 == 256 threads exactly
  const int wcol = tid >> 2;
  const int wkk  = (tid & 3) << 3;

  for (int k0 = 0; k0 < cinp; k0 += 32) {
    // stage A tile: 128 rows x 32 cols, float4 chunks (1024 chunks, 4/thread)
    #pragma unroll
    for (int it = 0; it < 4; ++it) {
      int idx = tid + (it << 8);
      int r  = idx >> 3;
      int c4 = (idx & 7) << 2;
      int grow = row0 + r;
      float4 f = make_float4(0.f, 0.f, 0.f, 0.f);
      const float* ap = &A[(size_t)grow * cinp + k0 + c4];
      if (grow < N) f = *(const float4*)ap;
      if (k0 + 32 < cinp) __builtin_prefetch(ap + 32, 0, 1);  // speculative OK
      uint2 p;
      p.x = (unsigned)bfbits(f.x) | ((unsigned)bfbits(f.y) << 16);
      p.y = (unsigned)bfbits(f.z) | ((unsigned)bfbits(f.w) << 16);
      *(uint2*)&sA[r * LDA + c4] = p;
    }
    // stage W tile: contiguous 16B copy (pack already [col][cinp] per term)
    *(v8bf*)&sW[wcol * LDW + wkk] = *(const v8bf*)&W[(size_t)wcol * cinp + k0 + wkk];
    __syncthreads();

    union Frag { v16bf v; v8bf h[2]; };
    Frag fa;
    fa.h[0] = *(const v8bf*)&sA[m * LDA + base];
    fa.h[1] = *(const v8bf*)&sA[m * LDA + 16 + base];

    #pragma unroll
    for (int t = 0; t < 4; ++t) {
      int col = (t << 4) + (lane & 15);
      Frag fb;
      fb.h[0] = *(const v8bf*)&sW[col * LDW + base];
      fb.h[1] = *(const v8bf*)&sW[col * LDW + 16 + base];
      acc[t] = __builtin_amdgcn_wmma_f32_16x16x32_bf16(
          false, fa.v, false, fb.v, (short)0, acc[t], false, false);
    }
    __syncthreads();
  }

  // epilogue: C layout VGPR r -> (M = r + 8*(lane>=16), N = lane&15)
  const int rowbase = row0 + (wave << 4) + ((lane >= 16) ? 8 : 0);
  const int n15 = lane & 15;
  #pragma unroll
  for (int t = 0; t < 4; ++t) {
    int gcol = (t << 4) + n15;
    #pragma unroll
    for (int r = 0; r < 8; ++r) {
      int grow = rowbase + r;
      if (grow < N) {
        float val = acc[t][r];
        if (accumulate) val += H[(size_t)grow * HIDDEN + gcol];
        else if (bias)  val += bias[gcol];
        H[(size_t)grow * HIDDEN + gcol] = val;
      }
    }
  }
}

// ------------------------------ host orchestration -------------------------

namespace {

struct Ctx {
  hipStream_t st;
  int N, E;
  const int *src, *dst;
  const float *do_inv, *di_inv;
  float *s, *t;   // [N x 128] scratch
};

inline dim3 blks(long long n, int bs) { return dim3((unsigned)((n + bs - 1) / bs)); }

void launch_zero(hipStream_t st, float* p, long long n) {
  zero_kernel<<<blks(n, 256), 256, 0, st>>>(p, n);
}

void gemm_call(const Ctx& c, float* H, const float* A, const bf16_t* W,
               const float* bias, int cinp, int acc) {
  gemm_wmma<<<dim3((c.N + GEMM_ROWS - 1) / GEMM_ROWS), 256, 0, c.st>>>(
      A, W, H, bias, c.N, cinp, acc);
}

void spmm_call(const Ctx& c, const float* sbuf, const int* from, const int* to,
               float* out, int cin, int cinp) {
  launch_zero(c.st, out, (long long)c.N * cinp);
  long long total = (long long)c.E * (cin >> 1);
  spmm_kernel<<<blks(total, 256), 256, 0, c.st>>>(sbuf, from, to, out, cin, cinp, total);
}

// DConv for one or two gates sharing the same input v (Z and R share diffusion).
void dconv(const Ctx& c, const float* v, int cin, int cinp,
           const bf16_t* W0, const float* b0, float* H0,
           const bf16_t* W1, const float* b1, float* H1) {
  const size_t term = (size_t)64 * cinp;
  long long nc = (long long)c.N * cinp;
  // term 0: v @ (W[0,0]+W[1,0]) + bias
  gemm_call(c, H0, v, W0 + 0 * term, b0, cinp, 0);
  if (W1) gemm_call(c, H1, v, W1 + 0 * term, b1, cinp, 0);
  // out-direction: t1o = pout(v)
  scale_kernel<<<blks(nc, 256), 256, 0, c.st>>>(c.s, v, c.do_inv, nc, cinp);
  spmm_call(c, c.s, c.src, c.dst, c.t, cin, cinp);
  gemm_call(c, H0, c.t, W0 + 1 * term, nullptr, cinp, 1);
  if (W1) gemm_call(c, H1, c.t, W1 + 1 * term, nullptr, cinp, 1);
  // t2o = 2*pout(t1o) - v
  scale_kernel<<<blks(nc, 256), 256, 0, c.st>>>(c.s, c.t, c.do_inv, nc, cinp);
  spmm_call(c, c.s, c.src, c.dst, c.t, cin, cinp);
  axpby_kernel<<<blks(nc, 256), 256, 0, c.st>>>(c.t, v, nc);
  gemm_call(c, H0, c.t, W0 + 3 * term, nullptr, cinp, 1);
  if (W1) gemm_call(c, H1, c.t, W1 + 3 * term, nullptr, cinp, 1);
  // in-direction: t1i = pin(v)
  scale_kernel<<<blks(nc, 256), 256, 0, c.st>>>(c.s, v, c.di_inv, nc, cinp);
  spmm_call(c, c.s, c.dst, c.src, c.t, cin, cinp);
  gemm_call(c, H0, c.t, W0 + 2 * term, nullptr, cinp, 1);
  if (W1) gemm_call(c, H1, c.t, W1 + 2 * term, nullptr, cinp, 1);
  // t2i = 2*pin(t1i) - v
  scale_kernel<<<blks(nc, 256), 256, 0, c.st>>>(c.s, c.t, c.di_inv, nc, cinp);
  spmm_call(c, c.s, c.dst, c.src, c.t, cin, cinp);
  axpby_kernel<<<blks(nc, 256), 256, 0, c.st>>>(c.t, v, nc);
  gemm_call(c, H0, c.t, W0 + 4 * term, nullptr, cinp, 1);
  if (W1) gemm_call(c, H1, c.t, W1 + 4 * term, nullptr, cinp, 1);
}

// One DCRNN cell. h0 may be nullptr (zeros).
void run_cell(const Ctx& c, const float* xin, int F, const float* h0,
              const bf16_t* Wpack, const float* bias,
              float* v, float* H, float* Z, float* h_out) {
  int cin  = F + HIDDEN;
  int cinp = (cin + 31) & ~31;
  size_t gate = (size_t)5 * 64 * cinp;
  long long nc  = (long long)c.N * cinp;
  long long nh  = (long long)c.N * HIDDEN;
  // v = [x | h0 | 0]; Z/R gates share diffusion terms
  concat_kernel<<<blks(nc, 256), 256, 0, c.st>>>(v, xin, F, h0, nullptr, nc, cinp);
  dconv(c, v, cin, cinp, Wpack + 0 * gate, bias + 0, Z,
                         Wpack + 1 * gate, bias + 64, H);
  sigmoid_kernel<<<blks(nh, 256), 256, 0, c.st>>>(Z, nh);
  // candidate input: [x | sigmoid(Hr) * h0 | 0]  (R never materialized)
  concat_kernel<<<blks(nc, 256), 256, 0, c.st>>>(v, xin, F, h0, H, nc, cinp);
  dconv(c, v, cin, cinp, Wpack + 2 * gate, bias + 128, H, nullptr, nullptr, nullptr);
  gru_kernel<<<blks(nh, 256), 256, 0, c.st>>>(h_out, Z, h0, H, nh);
}

} // namespace

extern "C" void kernel_launch(void* const* d_in, const int* in_sizes, int n_in,
                              void* d_out, int out_size, void* d_ws, size_t ws_size,
                              hipStream_t stream) {
  const float* x      = (const float*)d_in[0];
  const int*   eidx   = (const int*)d_in[1];
  const float* ew     = (const float*)d_in[2];
  const float* enc1_W = (const float*)d_in[3];
  const float* enc1_b = (const float*)d_in[4];
  const float* enc2_W = (const float*)d_in[5];
  const float* enc2_b = (const float*)d_in[6];
  const float* dec1_W = (const float*)d_in[7];
  const float* dec1_b = (const float*)d_in[8];
  const float* dec2_W = (const float*)d_in[9];
  const float* dec2_b = (const float*)d_in[10];
  const float* lin_W  = (const float*)d_in[11];
  const float* lin_b  = (const float*)d_in[12];
  float* out = (float*)d_out;

  const int N = NN;
  const int E = in_sizes[1] / 2;
  const int* src = eidx;
  const int* dst = eidx + E;

  // ---- workspace carve (all offsets 256B aligned) ----
  char* basep = (char*)d_ws;
  size_t off = 0;
  auto alloc = [&](size_t bytes) -> void* {
    void* p = basep + off;
    off = (off + bytes + 255) & ~(size_t)255;
    return p;
  };
  float* do_inv = (float*)alloc((size_t)N * 4);
  float* di_inv = (float*)alloc((size_t)N * 4);
  float* xs     = (float*)alloc((size_t)N * 2 * 4);
  float* Hbuf   = (float*)alloc((size_t)N * HIDDEN * 4);
  float* Zbuf   = (float*)alloc((size_t)N * HIDDEN * 4);
  float* hE1    = (float*)alloc((size_t)N * HIDDEN * 4);
  float* hE2    = (float*)alloc((size_t)N * HIDDEN * 4);
  float* hr     = (float*)alloc((size_t)N * HIDDEN * 4);
  float* vbuf   = (float*)alloc((size_t)N * 128 * 4);
  float* sbuf   = (float*)alloc((size_t)N * 128 * 4);
  float* tbuf   = (float*)alloc((size_t)N * 128 * 4);
  const int cin1  = 2 + HIDDEN;                 // 66
  const int cin2  = HIDDEN + HIDDEN;            // 128
  const int cinp1 = (cin1 + 31) & ~31;          // 96
  const int cinp2 = (cin2 + 31) & ~31;          // 128
  bf16_t* pE1 = (bf16_t*)alloc((size_t)3 * 5 * 64 * cinp1 * 2);
  bf16_t* pE2 = (bf16_t*)alloc((size_t)3 * 5 * 64 * cinp2 * 2);
  bf16_t* pD1 = (bf16_t*)alloc((size_t)3 * 5 * 64 * cinp1 * 2);
  bf16_t* pD2 = (bf16_t*)alloc((size_t)3 * 5 * 64 * cinp2 * 2);
  (void)ws_size; (void)n_in; (void)out_size;

  // ---- degrees + normalization ----
  launch_zero(stream, do_inv, N);
  launch_zero(stream, di_inv, N);
  degree_kernel<<<blks(E, 256), 256, 0, stream>>>(ew, src, dst, do_inv, di_inv, E);
  invert_kernel<<<blks(N, 256), 256, 0, stream>>>(do_inv, N);
  invert_kernel<<<blks(N, 256), 256, 0, stream>>>(di_inv, N);

  // ---- xs = x[:, :, 0, :] ----
  extract_xs_kernel<<<blks((long long)N * 2, 256), 256, 0, stream>>>(x, xs, (long long)N * 2);

  // ---- repack weights to bf16 term blocks (col-major per term, k padded) ----
  {
    long long t1 = 3LL * 5 * 64 * cinp1, t2 = 3LL * 5 * 64 * cinp2;
    pack_weights_kernel<<<blks(t1, 256), 256, 0, stream>>>(enc1_W, pE1, cin1, cinp1, t1);
    pack_weights_kernel<<<blks(t2, 256), 256, 0, stream>>>(enc2_W, pE2, cin2, cinp2, t2);
    pack_weights_kernel<<<blks(t1, 256), 256, 0, stream>>>(dec1_W, pD1, cin1, cinp1, t1);
    pack_weights_kernel<<<blks(t2, 256), 256, 0, stream>>>(dec2_W, pD2, cin2, cinp2, t2);
  }

  Ctx c{stream, N, E, src, dst, do_inv, di_inv, sbuf, tbuf};
  long long nh = (long long)N * HIDDEN;

  // enc1: h_enc1 = cell(xs, 0)
  run_cell(c, xs, 2, nullptr, pE1, enc1_b, vbuf, Hbuf, Zbuf, hE1);
  // h = relu(h_enc1)
  relu_kernel<<<blks(nh, 256), 256, 0, stream>>>(hE1, hr, nh);
  // enc2: h_enc2 = cell(h, 0)
  run_cell(c, hr, HIDDEN, nullptr, pE2, enc2_b, vbuf, Hbuf, Zbuf, hE2);
  // dec1: h = cell(xs, h_enc1)  -> hr, then relu in place
  run_cell(c, xs, 2, hE1, pD1, dec1_b, vbuf, Hbuf, Zbuf, hr);
  relu_kernel<<<blks(nh, 256), 256, 0, stream>>>(hr, hr, nh);
  // dec2: h = cell(h, h_enc2) -> reuse hE1
  run_cell(c, hr, HIDDEN, hE2, pD2, dec2_b, vbuf, Hbuf, Zbuf, hE1);

  // final linear [N,64] @ [64,2] + b
  linear_kernel<<<blks(N, 256), 256, 0, stream>>>(hE1, lin_W, lin_b, out, N);
}
